// AFWM_28956669509914
// MI455X (gfx1250) — matrix operand order repaired and checked by reading, not compiled, not run on hardware
//
#include <hip/hip_runtime.h>
#include <hip/hip_bf16.h>

// ---------------- CDNA5 WMMA types ----------------
typedef __attribute__((ext_vector_type(16))) __bf16 v16bf;
typedef __attribute__((ext_vector_type(8)))  float  v8f;

// ---------------- small device helpers ----------------
__device__ __forceinline__ float bf2f(unsigned short s) {
  unsigned int u = ((unsigned int)s) << 16;
  return __builtin_bit_cast(float, u);
}
__device__ __forceinline__ unsigned short f2bf(float f) {
  unsigned int u = __builtin_bit_cast(unsigned int, f);
  unsigned int r = u + 0x7fffu + ((u >> 16) & 1u);   // RNE truncate
  return (unsigned short)(r >> 16);
}
__device__ __forceinline__ __bf16 us2bf(unsigned short s) {
  return __builtin_bit_cast(__bf16, s);
}
__device__ __forceinline__ int clampi(int v, int lo, int hi) {
  return v < lo ? lo : (v > hi ? hi : v);
}

// =====================================================================
// Implicit-GEMM convolution via v_wmma_f32_16x16x32_bf16.
//   GEMM: D[M=Cout, N=B*Ho*Wo] = W[M,K=Cin*KH*KW] * Im2col[K,N]
//   Block: 128 threads = 4 wave32. Workgroup tile: 16(M) x 64(N),
//   each wave owns one 16x16 tile; A (weights) shared via LDS.
// =====================================================================
__global__ __launch_bounds__(128)
void wmma_conv_kernel(const unsigned short* __restrict__ X,   // bf16 [B,Cin,H,W]
                      const float* __restrict__ Wt,           // f32  [Cout,Cin,KH,KW]
                      const float* __restrict__ bias,         // f32  [Cout] or null
                      const unsigned short* __restrict__ Res, // bf16 residual or null
                      void* __restrict__ Y,                   // bf16 or f32 [B,Cout,Ho,Wo]
                      int Bn, int Cin, int H, int Wd,
                      int Cout, int KH, int KW, int stride, int pad,
                      int act, int out_f32)
{
  const int Ho = (H + 2 * pad - KH) / stride + 1;
  const int Wo = (Wd + 2 * pad - KW) / stride + 1;
  const long long Ntot = (long long)Bn * Ho * Wo;
  const int K = Cin * KH * KW;

  const int tid  = threadIdx.x;
  const int wave = tid >> 5;
  const int lane = tid & 31;
  const int mbase = blockIdx.y * 16;
  const long long nbase = (long long)blockIdx.x * 64;

  __shared__ __align__(32) unsigned short Alds[16 * 32];  // [m][k]  weights tile
  __shared__ __align__(32) unsigned short Blds[64 * 32];  // [n][k]  im2col tile (N-major)

  v8f acc = {0.f, 0.f, 0.f, 0.f, 0.f, 0.f, 0.f, 0.f};

  const int khkw = KH * KW;

  for (int kb = 0; kb < K; kb += 32) {
    // ---- cooperative A load: 16x32 weights, f32 -> bf16 ----
    for (int i = tid; i < 16 * 32; i += 128) {
      int m = i >> 5, kk = i & 31, k = kb + kk;
      float v = 0.f;
      int mg = mbase + m;
      if (mg < Cout && k < K) v = Wt[(long long)mg * K + k];
      Alds[i] = f2bf(v);
    }
    // ---- cooperative B load: 64x32 im2col patches (bf16 passthrough) ----
    for (int i = tid; i < 64 * 32; i += 128) {
      int nl = i >> 5, kk = i & 31, k = kb + kk;
      long long ng = nbase + nl;
      unsigned short v = 0;
      if (ng < Ntot && k < K) {
        int wo = (int)(ng % Wo);
        long long t = ng / Wo;
        int ho = (int)(t % Ho);
        int b  = (int)(t / Ho);
        int c  = k / khkw;
        int rs = k - c * khkw;
        int r  = rs / KW;
        int s  = rs - r * KW;
        int hi = ho * stride - pad + r;
        int wi = wo * stride - pad + s;
        if (hi >= 0 && hi < H && wi >= 0 && wi < Wd)
          v = X[(((long long)b * Cin + c) * H + hi) * Wd + wi];
      }
      Blds[i] = v;
    }
    // gfx1250 prefetch of next weight tile (global_prefetch_b8)
    if (kb + 32 < K && tid < 16 && (mbase + tid) < Cout)
      __builtin_prefetch(&Wt[(long long)(mbase + tid) * K + kb + 32], 0, 3);

    __syncthreads();

    // ---- A fragment: lane<16 -> M=lane, K 0..15; lane>=16 -> K 16..31 ----
    const int kb0  = (lane >= 16) ? 16 : 0;
    const int mrow = lane & 15;
    v16bf af;
#pragma unroll
    for (int i = 0; i < 8; ++i) {
      af[i]     = us2bf(Alds[mrow * 32 + kb0 + i]);
      af[i + 8] = us2bf(Alds[mrow * 32 + kb0 + 8 + i]);
    }
    // ---- B fragment: lane<16 -> N=lane, K 0..15; lane>=16 -> K 16..31 ----
    const int nl   = wave * 16 + (lane & 15);
    const int kb16 = (lane >= 16) ? 16 : 0;
    v16bf bfr = *(const v16bf*)&Blds[nl * 32 + kb16];

    acc = __builtin_amdgcn_wmma_f32_16x16x32_bf16(
        false, af, false, bfr, (short)0, acc, false, false);

    __syncthreads();
  }

  // ---- epilogue: D layout VGPR r -> M=r+8*(lane>=16), N = lane&15 ----
  long long ng = nbase + wave * 16 + (lane & 15);
  if (ng < Ntot) {
    int wo = (int)(ng % Wo);
    long long t = ng / Wo;
    int ho = (int)(t % Ho);
    int b  = (int)(t / Ho);
#pragma unroll
    for (int r = 0; r < 8; ++r) {
      int mg = mbase + r + 8 * (lane >> 4);
      if (mg >= Cout) continue;
      float v = acc[r];
      if (bias) v += bias[mg];
      if (act == 1)      v = v > 0.f ? v : 0.f;        // relu
      else if (act == 2) v = v > 0.f ? v : 0.1f * v;   // leaky 0.1
      long long idx = (((long long)b * Cout + mg) * Ho + ho) * Wo + wo;
      if (Res) v += bf2f(Res[idx]);
      if (out_f32) ((float*)Y)[idx] = v;
      else         ((unsigned short*)Y)[idx] = f2bf(v);
    }
  }
}

// =====================================================================
// BatchNorm statistics (per channel over B,H,W) + apply with ReLU.
// =====================================================================
__global__ void chan_stats_kernel(const unsigned short* __restrict__ X,
                                  float* __restrict__ mean, float* __restrict__ var,
                                  int Bn, int C, int HW)
{
  int c = blockIdx.x;
  __shared__ float ssum[256], ssq[256];
  float s = 0.f, q = 0.f;
  long long total = (long long)Bn * HW;
  for (long long i = threadIdx.x; i < total; i += blockDim.x) {
    int b  = (int)(i / HW);
    int hw = (int)(i % HW);
    float v = bf2f(X[((long long)b * C + c) * HW + hw]);
    s += v; q += v * v;
  }
  ssum[threadIdx.x] = s; ssq[threadIdx.x] = q;
  __syncthreads();
  for (int o = 128; o > 0; o >>= 1) {
    if ((int)threadIdx.x < o) {
      ssum[threadIdx.x] += ssum[threadIdx.x + o];
      ssq[threadIdx.x]  += ssq[threadIdx.x + o];
    }
    __syncthreads();
  }
  if (threadIdx.x == 0) {
    float m = ssum[0] / (float)total;
    mean[c] = m;
    var[c]  = ssq[0] / (float)total - m * m;
  }
}

__global__ void bn_act_kernel(const unsigned short* __restrict__ X,
                              const float* __restrict__ mean, const float* __restrict__ var,
                              const float* __restrict__ g, const float* __restrict__ b,
                              unsigned short* __restrict__ Y,
                              int C, int HW, long long total)
{
  long long i = (long long)blockIdx.x * blockDim.x + threadIdx.x;
  if (i >= total) return;
  int c = (int)((i / HW) % C);
  float v = bf2f(X[i]);
  v = (v - mean[c]) * rsqrtf(var[c] + 1e-5f) * g[c] + b[c];
  v = v > 0.f ? v : 0.f;   // reference always pairs BN with ReLU before conv
  Y[i] = f2bf(v);
}

// =====================================================================
// dtype converts / zero fill
// =====================================================================
__global__ void f2b_kernel(const float* __restrict__ X, unsigned short* __restrict__ Y, long long n)
{
  long long i = (long long)blockIdx.x * blockDim.x + threadIdx.x;
  if (i < n) Y[i] = f2bf(X[i]);
}
__global__ void b2f_kernel(const unsigned short* __restrict__ X, float* __restrict__ Y, long long n)
{
  long long i = (long long)blockIdx.x * blockDim.x + threadIdx.x;
  if (i < n) Y[i] = bf2f(X[i]);
}
__global__ void zero_kernel(float* __restrict__ Y, long long n)
{
  long long i = (long long)blockIdx.x * blockDim.x + threadIdx.x;
  if (i < n) Y[i] = 0.f;
}

// =====================================================================
// FPN nearest-2x upsample-add (in place on dst)
// =====================================================================
__global__ void up2add_kernel(unsigned short* __restrict__ dst,
                              const unsigned short* __restrict__ src,
                              int Bn, int C, int h, int w)
{
  long long total = (long long)Bn * C * 4 * h * w;
  long long i = (long long)blockIdx.x * blockDim.x + threadIdx.x;
  if (i >= total) return;
  int W2 = 2 * w, H2 = 2 * h;
  int x = (int)(i % W2);
  long long t = i / W2;
  int y = (int)(t % H2); t /= H2;
  int c = (int)(t % C);
  int b = (int)(t / C);
  float v = bf2f(dst[i]) + bf2f(src[(((long long)b * C + c) * h + (y >> 1)) * w + (x >> 1)]);
  dst[i] = f2bf(v);
}

// =====================================================================
// 7x7 correlation (mean over C) + leaky relu 0.1
// =====================================================================
__global__ void corr_kernel(const unsigned short* __restrict__ A,
                            const unsigned short* __restrict__ Bm,
                            unsigned short* __restrict__ O,
                            int Bn, int C, int H, int W)
{
  long long total = (long long)Bn * 49 * H * W;
  long long i = (long long)blockIdx.x * blockDim.x + threadIdx.x;
  if (i >= total) return;
  int x = (int)(i % W);
  long long t = i / W;
  int y  = (int)(t % H); t /= H;
  int tp = (int)(t % 49);
  int b  = (int)(t / 49);
  int dy = tp / 7 - 3, dx = tp % 7 - 3;
  int yy = y + dy, xx = x + dx;
  float s = 0.f;
  if (yy >= 0 && yy < H && xx >= 0 && xx < W) {
    for (int c = 0; c < C; ++c)
      s += bf2f(A[(((long long)b * C + c) * H + y) * W + x]) *
           bf2f(Bm[(((long long)b * C + c) * H + yy) * W + xx]);
  }
  s /= (float)C;
  s = s > 0.f ? s : 0.1f * s;
  O[i] = f2bf(s);
}

// =====================================================================
// Bilinear grid_sample. grid: f32 [B,2,Hg,Wg], normalized xy.
// =====================================================================
__global__ void grid_sample_kernel(const void* __restrict__ img, int img_f32,
                                   int C, int H, int W,
                                   const float* __restrict__ grid,
                                   int Hg, int Wg, int zero_pad,
                                   void* __restrict__ out, int out_f32, int Bn)
{
  long long total = (long long)Bn * C * Hg * Wg;
  long long i = (long long)blockIdx.x * blockDim.x + threadIdx.x;
  if (i >= total) return;
  int xg = (int)(i % Wg);
  long long t = i / Wg;
  int yg = (int)(t % Hg); t /= Hg;
  int c  = (int)(t % C);
  int b  = (int)(t / C);
  long long hw = (long long)Hg * Wg;
  long long gb = (long long)b * 2 * hw + (long long)yg * Wg + xg;
  float gx = grid[gb];
  float gy = grid[gb + hw];
  float x = (gx + 1.f) * (W * 0.5f) - 0.5f;
  float y = (gy + 1.f) * (H * 0.5f) - 0.5f;
  if (!zero_pad) {
    x = fminf(fmaxf(x, 0.f), W - 1.f);
    y = fminf(fmaxf(y, 0.f), H - 1.f);
  }
  float xf = floorf(x), yf = floorf(y);
  int x0 = (int)xf, y0 = (int)yf;
  float wx = x - xf, wy = y - yf;
  auto fetch = [&](int yi, int xi) -> float {
    int yc = clampi(yi, 0, H - 1), xc = clampi(xi, 0, W - 1);
    long long idx = (((long long)b * C + c) * H + yc) * W + xc;
    float v = img_f32 ? ((const float*)img)[idx]
                      : bf2f(((const unsigned short*)img)[idx]);
    if (zero_pad && (yi < 0 || yi >= H || xi < 0 || xi >= W)) v = 0.f;
    return v;
  };
  float v = fetch(y0, x0) * (1 - wx) * (1 - wy) + fetch(y0, x0 + 1) * wx * (1 - wy) +
            fetch(y0 + 1, x0) * (1 - wx) * wy + fetch(y0 + 1, x0 + 1) * wx * wy;
  if (out_f32) ((float*)out)[i] = v;
  else         ((unsigned short*)out)[i] = f2bf(v);
}

// =====================================================================
// apply_offset: offset f32 [B,2,H,W] -> normalized grid (align_corners=True)
// =====================================================================
__global__ void apply_offset_kernel(const float* __restrict__ off,
                                    float* __restrict__ grid,
                                    int Bn, int H, int W)
{
  long long total = (long long)Bn * H * W;
  long long i = (long long)blockIdx.x * blockDim.x + threadIdx.x;
  if (i >= total) return;
  int x = (int)(i % W);
  long long t = i / W;
  int y = (int)(t % H);
  int b = (int)(t / H);
  long long hw = (long long)H * W;
  long long base = (long long)b * 2 * hw + (long long)y * W + x;
  float gx = x + off[base];
  float gy = y + off[base + hw];
  grid[base]      = gx / ((W - 1.f) * 0.5f) - 1.f;
  grid[base + hw] = gy / ((H - 1.f) * 0.5f) - 1.f;
}

// =====================================================================
// Bilinear resize (half-pixel centers), dtype-flagged
// =====================================================================
__global__ void resize_bl_kernel(const void* __restrict__ X, int xf32,
                                 int C, int H, int W,
                                 void* __restrict__ Y, int of32,
                                 int Ho, int Wo, int Bn)
{
  long long total = (long long)Bn * C * Ho * Wo;
  long long i = (long long)blockIdx.x * blockDim.x + threadIdx.x;
  if (i >= total) return;
  int x = (int)(i % Wo);
  long long t = i / Wo;
  int y = (int)(t % Ho); t /= Ho;
  int c = (int)(t % C);
  int b = (int)(t / C);
  float sx = (float)W / Wo, sy = (float)H / Ho;
  float xi = (x + 0.5f) * sx - 0.5f;
  float yi = (y + 0.5f) * sy - 0.5f;
  xi = fminf(fmaxf(xi, 0.f), W - 1.f);
  yi = fminf(fmaxf(yi, 0.f), H - 1.f);
  int x0 = (int)xi, y0 = (int)yi;
  int x1 = clampi(x0 + 1, 0, W - 1), y1 = clampi(y0 + 1, 0, H - 1);
  float wx = xi - x0, wy = yi - y0;
  auto at = [&](int yy, int xx) -> float {
    long long idx = (((long long)b * C + c) * H + yy) * W + xx;
    return xf32 ? ((const float*)X)[idx] : bf2f(((const unsigned short*)X)[idx]);
  };
  float v = at(y0, x0) * (1 - wx) * (1 - wy) + at(y0, x1) * wx * (1 - wy) +
            at(y1, x0) * (1 - wx) * wy + at(y1, x1) * wx * wy;
  if (of32) ((float*)Y)[i] = v;
  else      ((unsigned short*)Y)[i] = f2bf(v);
}

// =====================================================================
// channel concat (2 tensors, NCHW)
// =====================================================================
__global__ void concat2_kernel(const unsigned short* __restrict__ A,
                               const unsigned short* __restrict__ Bt,
                               unsigned short* __restrict__ O,
                               int Bn, int C1, int C2, int HW)
{
  long long total = (long long)Bn * (C1 + C2) * HW;
  long long i = (long long)blockIdx.x * blockDim.x + threadIdx.x;
  if (i >= total) return;
  int hw = (int)(i % HW);
  long long t = i / HW;
  int c = (int)(t % (C1 + C2));
  int b = (int)(t / (C1 + C2));
  O[i] = (c < C1) ? A[((long long)b * C1 + c) * HW + hw]
                  : Bt[((long long)b * C2 + (c - C1)) * HW + hw];
}

// =====================================================================
// DELTA_W conv (4 fixed 3x3 filters, pad 0) on one flow channel
// =====================================================================
__device__ __constant__ float DWF[4][3][3] = {
  {{0, 0, 0}, {1, -2, 1}, {0, 0, 0}},
  {{0, 1, 0}, {0, -2, 0}, {0, 1, 0}},
  {{1, 0, 0}, {0, -2, 0}, {0, 0, 1}},
  {{0, 0, 1}, {0, -2, 0}, {1, 0, 0}}};

__global__ void delta_conv_kernel(const float* __restrict__ F, int ch, int Cflow,
                                  float* __restrict__ O, int Bn, int H, int W)
{
  int Ho = H - 2, Wo = W - 2;
  long long total = (long long)Bn * 4 * Ho * Wo;
  long long i = (long long)blockIdx.x * blockDim.x + threadIdx.x;
  if (i >= total) return;
  int x = (int)(i % Wo);
  long long t = i / Wo;
  int y = (int)(t % Ho); t /= Ho;
  int f = (int)(t % 4);
  int b = (int)(t / 4);
  float s = 0.f;
#pragma unroll
  for (int r = 0; r < 3; ++r)
#pragma unroll
    for (int c = 0; c < 3; ++c)
      s += DWF[f][r][c] * F[(((long long)b * Cflow + ch) * H + y + r) * W + x + c];
  O[i] = s;
}

// =====================================================================
// Host orchestration
// =====================================================================
static inline long long cdivll(long long a, long long b) { return (a + b - 1) / b; }

extern "C" void kernel_launch(void* const* d_in, const int* in_sizes, int n_in,
                              void* d_out, int out_size, void* d_ws, size_t ws_size,
                              hipStream_t stream)
{
  const int Bn = 8, Hin = 256, Win = 192;
  const int CH[5] = {64, 128, 256, 256, 256};
  const int FPND = 256;
  int Hs[5], Wss[5];
  { int h = Hin, w = Win; for (int i = 0; i < 5; i++) { h /= 2; w /= 2; Hs[i] = h; Wss[i] = w; } }

  const float* cond_in = (const float*)d_in[0];
  const float* img_in  = (const float*)d_in[1];
  const float* edge_in = (const float*)d_in[2];

  int cur = 3;
  auto nf = [&]() -> const float* {
    int idx = cur < n_in ? cur : (n_in - 1);
    ++cur;
    return (const float*)d_in[idx];
  };

  struct RB { const float *g1, *b1, *w1, *g2, *b2, *w2; };
  struct Stage { const float *g, *b, *w; RB r1, r2; };
  Stage cenc[5], ienc[5];
  auto rdRB = [&](RB& r) { r.g1 = nf(); r.b1 = nf(); r.w1 = nf(); r.g2 = nf(); r.b2 = nf(); r.w2 = nf(); };
  auto rdStage = [&](Stage& s) { s.g = nf(); s.b = nf(); s.w = nf(); rdRB(s.r1); rdRB(s.r2); };
  for (int i = 0; i < 5; i++) rdStage(cenc[i]);
  for (int i = 0; i < 5; i++) rdStage(ienc[i]);
  struct FpnP { const float *aw[5], *ab[5], *sw[5], *sb[5]; };
  FpnP cf, gf;
  auto rdF = [&](FpnP& f) {
    for (int i = 0; i < 5; i++) f.aw[i] = nf();
    for (int i = 0; i < 5; i++) f.ab[i] = nf();
    for (int i = 0; i < 5; i++) f.sw[i] = nf();
    for (int i = 0; i < 5; i++) f.sb[i] = nf();
  };
  rdF(cf); rdF(gf);
  const float *mw[5][4], *mb[5][4], *rw[5][4], *rbp[5][4];
  for (int i = 0; i < 5; i++) for (int j = 0; j < 4; j++) { mw[i][j] = nf(); mb[i][j] = nf(); }
  for (int i = 0; i < 5; i++) for (int j = 0; j < 4; j++) { rw[i][j] = nf(); rbp[i][j] = nf(); }

  // bump allocator over d_ws (wraps deterministically if exceeded)
  char* wsp = (char*)d_ws;
  size_t woff = 0;
  auto alloc = [&](size_t bytes) -> void* {
    size_t a = (bytes + 255) & ~(size_t)255;
    if (woff + a > ws_size) woff = 0;
    void* p = wsp + woff;
    woff += a;
    return p;
  };
  auto abf = [&](long long n) { return (unsigned short*)alloc((size_t)n * 2); };
  auto af32 = [&](long long n) { return (float*)alloc((size_t)n * 4); };

  auto conv = [&](const unsigned short* X, const float* Wt, const float* bias,
                  const unsigned short* Res, void* Y,
                  int Cin, int H, int W, int Cout, int KH, int KW,
                  int st, int pd, int act, int of32) {
    int Ho = (H + 2 * pd - KH) / st + 1;
    int Wo = (W + 2 * pd - KW) / st + 1;
    long long N = (long long)Bn * Ho * Wo;
    dim3 g((unsigned)cdivll(N, 64), (unsigned)cdivll(Cout, 16));
    wmma_conv_kernel<<<g, 128, 0, stream>>>(X, Wt, bias, Res, Y, Bn, Cin, H, W,
                                            Cout, KH, KW, st, pd, act, of32);
  };

  float* statm = af32(512);
  float* statv = af32(512);
  auto bnrelu = [&](const unsigned short* X, const float* g, const float* b,
                    unsigned short* Y, int C, int H, int W) {
    int HW = H * W;
    chan_stats_kernel<<<C, 256, 0, stream>>>(X, statm, statv, Bn, C, HW);
    long long tot = (long long)Bn * C * HW;
    bn_act_kernel<<<(unsigned)cdivll(tot, 256), 256, 0, stream>>>(X, statm, statv, g, b, Y, C, HW, tot);
  };

  // ---------------- encoders ----------------
  auto run_enc = [&](const float* inp, Stage* st, unsigned short** feats) {
    long long n0 = (long long)Bn * 3 * Hin * Win;
    unsigned short* x = abf(n0);
    f2b_kernel<<<(unsigned)cdivll(n0, 256), 256, 0, stream>>>(inp, x, n0);
    int C = 3, H = Hin, W = Win;
    for (int i = 0; i < 5; i++) {
      unsigned short* t = abf((long long)Bn * C * H * W);
      bnrelu(x, st[i].g, st[i].b, t, C, H, W);
      int Co = CH[i], Ho = Hs[i], Wo = Wss[i];
      unsigned short* y = abf((long long)Bn * Co * Ho * Wo);
      conv(t, st[i].w, nullptr, nullptr, y, C, H, W, Co, 3, 3, 2, 1, 0, 0);
      for (int r = 0; r < 2; r++) {
        RB& rb = r ? st[i].r2 : st[i].r1;
        unsigned short* t1 = abf((long long)Bn * Co * Ho * Wo);
        bnrelu(y, rb.g1, rb.b1, t1, Co, Ho, Wo);
        unsigned short* h = abf((long long)Bn * Co * Ho * Wo);
        conv(t1, rb.w1, nullptr, nullptr, h, Co, Ho, Wo, Co, 3, 3, 1, 1, 0, 0);
        unsigned short* t2 = abf((long long)Bn * Co * Ho * Wo);
        bnrelu(h, rb.g2, rb.b2, t2, Co, Ho, Wo);
        unsigned short* y2 = abf((long long)Bn * Co * Ho * Wo);
        conv(t2, rb.w2, nullptr, y /*residual*/, y2, Co, Ho, Wo, Co, 3, 3, 1, 1, 0, 0);
        y = y2;
      }
      feats[i] = y;
      x = y; C = Co; H = Ho; W = Wo;
    }
  };

  // ---------------- FPN ----------------
  auto run_fpn = [&](unsigned short** feats, FpnP& fp, unsigned short** pyr) {
    unsigned short* last = nullptr;
    int lh = 0, lw = 0;
    for (int i = 0; i < 5; i++) {
      int k = 4 - i;
      int H = Hs[k], W = Wss[k], C = CH[k];
      unsigned short* h = abf((long long)Bn * FPND * H * W);
      conv(feats[k], fp.aw[i], fp.ab[i], nullptr, h, C, H, W, FPND, 1, 1, 1, 0, 0, 0);
      if (last) {
        long long tot = (long long)Bn * FPND * H * W;
        up2add_kernel<<<(unsigned)cdivll(tot, 256), 256, 0, stream>>>(h, last, Bn, FPND, lh, lw);
      }
      unsigned short* o = abf((long long)Bn * FPND * H * W);
      conv(h, fp.sw[i], fp.sb[i], nullptr, o, FPND, H, W, FPND, 3, 3, 1, 1, 0, 0);
      last = o; lh = H; lw = W;
      pyr[k] = o;
    }
  };

  unsigned short *cfeat[5], *ifeat[5], *cpyr[5], *ipyr[5];
  run_enc(cond_in, cenc, cfeat);
  run_enc(img_in,  ienc, ifeat);
  run_fpn(cfeat, cf, cpyr);
  run_fpn(ifeat, gf, ipyr);

  // ---------------- flow cascade ----------------
  float* last_flow = nullptr;
  float* lf_all[5]; int lf_h[5], lf_w[5];
  float* deltas[10]; int d_h[10], d_w[10];
  float* xall[5]; int xa_h[5], xa_w[5];
  float* eall[5];
  float* dxall[5];
  float* dyall[5];

  for (int i = 0; i < 5; i++) {
    int k = 4 - i;
    int H = Hs[k], W = Wss[k];
    long long hw = (long long)H * W;
    unsigned short* xw = ipyr[k];
    unsigned short* xc = cpyr[k];
    unsigned short* xwa = xw;
    if (last_flow) {
      unsigned short* tmp = abf((long long)Bn * FPND * hw);
      long long tot = (long long)Bn * FPND * hw;
      grid_sample_kernel<<<(unsigned)cdivll(tot, 256), 256, 0, stream>>>(
          xw, 0, FPND, H, W, last_flow, H, W, 0, tmp, 0, Bn);
      xwa = tmp;
    }
    unsigned short* corr = abf((long long)Bn * 49 * hw);
    {
      long long tot = (long long)Bn * 49 * hw;
      corr_kernel<<<(unsigned)cdivll(tot, 256), 256, 0, stream>>>(xwa, xc, corr, Bn, FPND, H, W);
    }
    // main stack 49->128->64->32->2
    int cs[5] = {49, 128, 64, 32, 2};
    unsigned short* cu = corr;
    float* mflow = af32((long long)Bn * 2 * hw);
    for (int j = 0; j < 4; j++) {
      if (j < 3) {
        unsigned short* nx = abf((long long)Bn * cs[j + 1] * hw);
        conv(cu, mw[i][j], mb[i][j], nullptr, nx, cs[j], H, W, cs[j + 1], 3, 3, 1, 1, 2, 0);
        cu = nx;
      } else {
        conv(cu, mw[i][j], mb[i][j], nullptr, mflow, cs[j], H, W, 2, 3, 3, 1, 1, 0, 1);
      }
    }
    deltas[2 * i] = mflow; d_h[2 * i] = H; d_w[2 * i] = W;

    float* g = af32((long long)Bn * 2 * hw);
    {
      long long tot = (long long)Bn * hw;
      apply_offset_kernel<<<(unsigned)cdivll(tot, 256), 256, 0, stream>>>(mflow, g, Bn, H, W);
    }
    float* flow = af32((long long)Bn * 2 * hw);
    if (last_flow) {
      long long tot = (long long)Bn * 2 * hw;
      grid_sample_kernel<<<(unsigned)cdivll(tot, 256), 256, 0, stream>>>(
          last_flow, 1, 2, H, W, g, H, W, 0, flow, 1, Bn);
    } else {
      hipMemcpyAsync(flow, g, (size_t)Bn * 2 * hw * 4, hipMemcpyDeviceToDevice, stream);
    }
    last_flow = flow;

    unsigned short* xw2 = abf((long long)Bn * FPND * hw);
    {
      long long tot = (long long)Bn * FPND * hw;
      grid_sample_kernel<<<(unsigned)cdivll(tot, 256), 256, 0, stream>>>(
          xw, 0, FPND, H, W, flow, H, W, 0, xw2, 0, Bn);
    }
    unsigned short* cat = abf((long long)Bn * 512 * hw);
    {
      long long tot = (long long)Bn * 512 * hw;
      concat2_kernel<<<(unsigned)cdivll(tot, 256), 256, 0, stream>>>(
          xw2, xc, cat, Bn, FPND, FPND, (int)hw);
    }
    int rs[5] = {512, 128, 64, 32, 2};
    unsigned short* ru = cat;
    float* rflow = af32((long long)Bn * 2 * hw);
    for (int j = 0; j < 4; j++) {
      if (j < 3) {
        unsigned short* nx = abf((long long)Bn * rs[j + 1] * hw);
        conv(ru, rw[i][j], rbp[i][j], nullptr, nx, rs[j], H, W, rs[j + 1], 3, 3, 1, 1, 2, 0);
        ru = nx;
      } else {
        conv(ru, rw[i][j], rbp[i][j], nullptr, rflow, rs[j], H, W, 2, 3, 3, 1, 1, 0, 1);
      }
    }
    deltas[2 * i + 1] = rflow; d_h[2 * i + 1] = H; d_w[2 * i + 1] = W;

    float* g2 = af32((long long)Bn * 2 * hw);
    {
      long long tot = (long long)Bn * hw;
      apply_offset_kernel<<<(unsigned)cdivll(tot, 256), 256, 0, stream>>>(rflow, g2, Bn, H, W);
    }
    float* flow2 = af32((long long)Bn * 2 * hw);
    {
      long long tot = (long long)Bn * 2 * hw;
      grid_sample_kernel<<<(unsigned)cdivll(tot, 256), 256, 0, stream>>>(
          last_flow, 1, 2, H, W, g2, H, W, 0, flow2, 1, Bn);
    }
    int H2 = H * 2, W2 = W * 2;
    float* up = af32((long long)Bn * 2 * H2 * W2);
    {
      long long tot = (long long)Bn * 2 * H2 * W2;
      resize_bl_kernel<<<(unsigned)cdivll(tot, 256), 256, 0, stream>>>(
          flow2, 1, 2, H, W, up, 1, H2, W2, Bn);
    }
    last_flow = up;
    lf_all[i] = up; lf_h[i] = H2; lf_w[i] = W2;

    float* curx = af32((long long)Bn * 3 * H2 * W2);
    {
      long long tot = (long long)Bn * 3 * H2 * W2;
      resize_bl_kernel<<<(unsigned)cdivll(tot, 256), 256, 0, stream>>>(
          img_in, 1, 3, Hin, Win, curx, 1, H2, W2, Bn);
    }
    float* xs = af32((long long)Bn * 3 * H2 * W2);
    {
      long long tot = (long long)Bn * 3 * H2 * W2;
      grid_sample_kernel<<<(unsigned)cdivll(tot, 256), 256, 0, stream>>>(
          curx, 1, 3, H2, W2, up, H2, W2, 0, xs, 1, Bn);
    }
    xall[i] = xs; xa_h[i] = H2; xa_w[i] = W2;

    float* cure = af32((long long)Bn * H2 * W2);
    {
      long long tot = (long long)Bn * H2 * W2;
      resize_bl_kernel<<<(unsigned)cdivll(tot, 256), 256, 0, stream>>>(
          edge_in, 1, 1, Hin, Win, cure, 1, H2, W2, Bn);
    }
    float* es = af32((long long)Bn * H2 * W2);
    {
      long long tot = (long long)Bn * H2 * W2;
      grid_sample_kernel<<<(unsigned)cdivll(tot, 256), 256, 0, stream>>>(
          cure, 1, 1, H2, W2, up, H2, W2, 1, es, 1, Bn);
    }
    eall[i] = es;

    float* dx = af32((long long)Bn * 4 * (H2 - 2) * (W2 - 2));
    float* dy = af32((long long)Bn * 4 * (H2 - 2) * (W2 - 2));
    {
      long long tot = (long long)Bn * 4 * (H2 - 2) * (W2 - 2);
      delta_conv_kernel<<<(unsigned)cdivll(tot, 256), 256, 0, stream>>>(up, 0, 2, dx, Bn, H2, W2);
      delta_conv_kernel<<<(unsigned)cdivll(tot, 256), 256, 0, stream>>>(up, 1, 2, dy, Bn, H2, W2);
    }
    dxall[i] = dx; dyall[i] = dy;
  }

  float* xwarp = af32((long long)Bn * 3 * Hin * Win);
  {
    long long tot = (long long)Bn * 3 * Hin * Win;
    grid_sample_kernel<<<(unsigned)cdivll(tot, 256), 256, 0, stream>>>(
        img_in, 1, 3, Hin, Win, last_flow, Hin, Win, 0, xwarp, 1, Bn);
  }

  // ---------------- pack outputs (tuple order) ----------------
  zero_kernel<<<(unsigned)cdivll(out_size, 256), 256, 0, stream>>>((float*)d_out, out_size);
  float* outp = (float*)d_out;
  long long ooff = 0;
  auto packf = [&](const float* src, long long n) {
    if (ooff + n <= (long long)out_size)
      hipMemcpyAsync(outp + ooff, src, (size_t)n * 4, hipMemcpyDeviceToDevice, stream);
    ooff += n;
  };
  auto packb = [&](const unsigned short* src, long long n) {
    if (ooff + n <= (long long)out_size)
      b2f_kernel<<<(unsigned)cdivll(n, 256), 256, 0, stream>>>(src, outp + ooff, n);
    ooff += n;
  };
  packf(xwarp, (long long)Bn * 3 * Hin * Win);
  packf(last_flow, (long long)Bn * 2 * Hin * Win);
  for (int i = 0; i < 5; i++) { int k = 4 - i; packb(cpyr[k], (long long)Bn * FPND * Hs[k] * Wss[k]); }
  for (int i = 0; i < 5; i++) packf(lf_all[i], (long long)Bn * 2 * lf_h[i] * lf_w[i]);
  for (int i = 0; i < 10; i++) packf(deltas[i], (long long)Bn * 2 * d_h[i] * d_w[i]);
  for (int i = 0; i < 5; i++) packf(xall[i], (long long)Bn * 3 * xa_h[i] * xa_w[i]);
  for (int i = 0; i < 5; i++) packf(eall[i], (long long)Bn * xa_h[i] * xa_w[i]);
  for (int i = 0; i < 5; i++) packf(dxall[i], (long long)Bn * 4 * (xa_h[i] - 2) * (xa_w[i] - 2));
  for (int i = 0; i < 5; i++) packf(dyall[i], (long long)Bn * 4 * (xa_h[i] - 2) * (xa_w[i] - 2));
}
